// MultiHeadAttention_66460323938645
// MI455X (gfx1250) — matrix operand order repaired and checked by reading, not compiled, not run on hardware
//
#include <hip/hip_runtime.h>
#include <hip/hip_bf16.h>

// ---------------------------------------------------------------------------
// MHA forward for gfx1250: bf16 WMMA (16x16x32) everywhere, f32 accumulate.
// GEMM staging uses CDNA5 async global->LDS loads when the builtin exists.
// ---------------------------------------------------------------------------

typedef __attribute__((ext_vector_type(16))) __bf16 v16bf;
typedef __attribute__((ext_vector_type(8)))  __bf16 v8bf;
typedef __attribute__((ext_vector_type(8)))  float  v8f;

#define TSEQ 2048
#define CDIM 2048
#define NHEAD 16
#define HDIM 128
#define MROWS 4096            // B*T

#if defined(__AMDGCN__) && __has_builtin(__builtin_amdgcn_global_load_async_to_lds_b128)
#define HAVE_ASYNC_LDS 1
#endif

#if defined(HAVE_ASYNC_LDS)
typedef int v4i_vec __attribute__((vector_size(16)));
typedef __attribute__((address_space(1))) v4i_vec* g_v4i_ptr;
typedef __attribute__((address_space(3))) v4i_vec* l_v4i_ptr;
#endif

__device__ __forceinline__ v8f wmma_bf16(v16bf a, v16bf b, v8f c) {
  // D = A(16x32) * B(32x16) + C, f32 accum
  return __builtin_amdgcn_wmma_f32_16x16x32_bf16(false, a, false, b,
                                                 (short)0, c, false, false);
}

// Build a v16bf fragment from two 16-byte chunks.
__device__ __forceinline__ v16bf ld_frag(const __bf16* p0, const __bf16* p1) {
  v8bf lo = *(const v8bf*)p0;
  v8bf hi = *(const v8bf*)p1;
  v16bf r;
#pragma unroll
  for (int i = 0; i < 8; ++i) { r[i] = lo[i]; r[i + 8] = hi[i]; }
  return r;
}

// 16-byte global -> LDS copy: async DMA path on CDNA5, register round-trip else.
__device__ __forceinline__ void cp16_g2s(__bf16* lds_dst, const __bf16* gsrc) {
#if defined(HAVE_ASYNC_LDS)
  __builtin_amdgcn_global_load_async_to_lds_b128(
      (g_v4i_ptr)gsrc, (l_v4i_ptr)lds_dst, 0, 0);
#else
  *(v8bf*)lds_dst = *(const v8bf*)gsrc;
#endif
}

__device__ __forceinline__ void g2s_wait() {
#if defined(HAVE_ASYNC_LDS)
#if __has_builtin(__builtin_amdgcn_s_wait_asynccnt)
  __builtin_amdgcn_s_wait_asynccnt(0);
#else
  asm volatile("s_wait_asynccnt 0x0" ::: "memory");
#endif
#endif
}

__device__ __forceinline__ unsigned pack2_bf16(float a, float b) {
  unsigned short ua = __builtin_bit_cast(unsigned short, (__bf16)a);
  unsigned short ub = __builtin_bit_cast(unsigned short, (__bf16)b);
  return (unsigned)ua | ((unsigned)ub << 16);
}

// ---------------------------------------------------------------------------
// Elementwise f32 -> bf16
// ---------------------------------------------------------------------------
__global__ void cvt_bf16_kernel(const float* __restrict__ in,
                                __bf16* __restrict__ out, int n) {
  int i = blockIdx.x * 256 + threadIdx.x;
  if (i < n) out[i] = (__bf16)in[i];
}

// Tiled transpose + convert: in f32 [R][C] -> out bf16 [C][R]
__global__ void transpose_cvt_kernel(const float* __restrict__ in,
                                     __bf16* __restrict__ out, int R, int C) {
  __shared__ float tile[32][33];
  int c0 = blockIdx.x * 32, r0 = blockIdx.y * 32;
  int tx = threadIdx.x, ty = threadIdx.y;  // 32 x 8
#pragma unroll
  for (int j = ty; j < 32; j += 8)
    tile[j][tx] = in[(size_t)(r0 + j) * C + c0 + tx];
  __syncthreads();
#pragma unroll
  for (int j = ty; j < 32; j += 8)
    out[(size_t)(c0 + j) * R + r0 + tx] = (__bf16)tile[tx][j];
}

// ---------------------------------------------------------------------------
// GEMM: Cacc = A[M x K] * BT[N x K]^T   (A bf16 row-major, BT = B pre-transposed)
// mode 0: scatter into q[bh][t][hd], k[bh][t][hd], vT[bh][hd][t]  (bf16)
// mode 1: out[row][col] = acc + bias[col]                          (f32)
// Block tile 128(M) x 256(N), 8 waves (2x4), wave tile 64x64, K-step 32.
// ---------------------------------------------------------------------------
__global__ __launch_bounds__(256)
void gemm_bf16_kernel(const __bf16* __restrict__ A, const __bf16* __restrict__ BT,
                      int M, int N, int K, int mode,
                      __bf16* __restrict__ qb, __bf16* __restrict__ kbuf,
                      __bf16* __restrict__ vt,
                      const float* __restrict__ bias, float* __restrict__ out) {
  __shared__ __align__(16) __bf16 As[128 * 40];  // 128 A-rows, K=32, pad ->40
  __shared__ __align__(16) __bf16 Bs[256 * 40];  // 256 B-cols (rows of B^T)

  const int tid  = threadIdx.x;
  const int lane = tid & 31;
  const int wave = tid >> 5;
  const int hi   = lane >> 4;
  const int ln   = lane & 15;
  const int wm   = wave >> 2;      // 0..1 -> 64 rows each
  const int wn   = wave & 3;       // 0..3 -> 64 cols each
  const int m0   = blockIdx.y * 128;
  const int n0   = blockIdx.x * 256;

  const v8f vzero = {0.f, 0.f, 0.f, 0.f, 0.f, 0.f, 0.f, 0.f};
  v8f acc[4][4];
#pragma unroll
  for (int mi = 0; mi < 4; ++mi)
#pragma unroll
    for (int ni = 0; ni < 4; ++ni) acc[mi][ni] = vzero;

  // staging assignment:
  //  A tile 128x32: 2 threads/row, 16 bf16 each (2 x b128)
  //  B tile 256x32: 1 thread/row, 32 bf16 each (4 x b128)
  const int rA = tid >> 1;
  const int hA = (tid & 1) * 16;
  const __bf16* gA = A  + (size_t)(m0 + rA) * K + hA;
  const __bf16* gB = BT + (size_t)(n0 + tid) * K;
  __bf16* sAp = As + rA * 40 + hA;
  __bf16* sBp = Bs + tid * 40;

  const int kb_off = hi * 16;  // A-frag K base (0 or 16)
  const int koff   = hi * 16;  // B-frag K base (0 or 16)

  for (int k0 = 0; k0 < K; k0 += 32) {
    cp16_g2s(sAp,      gA + k0);
    cp16_g2s(sAp + 8,  gA + k0 + 8);
    cp16_g2s(sBp,      gB + k0);
    cp16_g2s(sBp + 8,  gB + k0 + 8);
    cp16_g2s(sBp + 16, gB + k0 + 16);
    cp16_g2s(sBp + 24, gB + k0 + 24);
    g2s_wait();
    __syncthreads();

    v16bf bfrag[4];
#pragma unroll
    for (int ni = 0; ni < 4; ++ni) {
      const __bf16* p = Bs + (wn * 64 + ni * 16 + ln) * 40 + koff;
      bfrag[ni] = ld_frag(p, p + 8);
    }
#pragma unroll
    for (int mi = 0; mi < 4; ++mi) {
      const __bf16* p = As + (wm * 64 + mi * 16 + ln) * 40 + kb_off;
      v16bf afrag = ld_frag(p, p + 8);
#pragma unroll
      for (int ni = 0; ni < 4; ++ni)
        acc[mi][ni] = wmma_bf16(afrag, bfrag[ni], acc[mi][ni]);
    }
    __syncthreads();
  }

  // epilogue: C layout -> element i is row (i + 8*hi), lane&15 is column
#pragma unroll
  for (int mi = 0; mi < 4; ++mi) {
#pragma unroll
    for (int ni = 0; ni < 4; ++ni) {
#pragma unroll
      for (int i = 0; i < 8; ++i) {
        int row = m0 + wm * 64 + mi * 16 + i + 8 * hi;
        int col = n0 + wn * 64 + ni * 16 + ln;
        float v = acc[mi][ni][i];
        if (mode == 0) {
          int b = row >> 11, t = row & (TSEQ - 1);
          int which = col >> 11, rem = col & (CDIM - 1);
          int h = rem >> 7, hd = rem & (HDIM - 1);
          int bh = b * NHEAD + h;
          if (which == 0)
            qb[((size_t)bh * TSEQ + t) * HDIM + hd] = (__bf16)v;
          else if (which == 1)
            kbuf[((size_t)bh * TSEQ + t) * HDIM + hd] = (__bf16)v;
          else
            vt[((size_t)bh * HDIM + hd) * TSEQ + t] = (__bf16)v;
        } else {
          out[(size_t)row * N + col] = v + bias[col];
        }
      }
    }
  }
}

// ---------------------------------------------------------------------------
// Flash attention, causal.  Grid: (T/128, B*H).  8 waves, 16 queries/wave.
// Scores computed transposed (S^T = K * Q^T) so the softmax row stats live one-
// per-lane and the O^T accumulator rescale is a uniform per-lane multiply.
// ---------------------------------------------------------------------------
__global__ __launch_bounds__(256)
void attn_kernel(const __bf16* __restrict__ qg, const __bf16* __restrict__ kg,
                 const __bf16* __restrict__ vtg, __bf16* __restrict__ yb) {
  __shared__ __align__(16) __bf16 pShared[8][16 * 40];  // per-wave P scratch

  const int lane = threadIdx.x & 31;
  const int wave = threadIdx.x >> 5;
  const int hi   = lane >> 4;
  const int ln   = lane & 15;
  const int bh   = blockIdx.y;
  const int q0   = blockIdx.x * 128 + wave * 16;  // this wave's query base

  const __bf16* qbh = qg  + (size_t)bh * TSEQ * HDIM;
  const __bf16* kbh = kg  + (size_t)bh * TSEQ * HDIM;
  const __bf16* vbh = vtg + (size_t)bh * HDIM * TSEQ;  // [d][t]
  __bf16* pl = &pShared[wave][0];

  const int kb_off = hi * 16;
  const int koff   = hi * 16;

  // persistent Q^T B-fragments: lane -> query q0+ln, element j -> hd = s*32+koff+j
  v16bf qf[4];
  {
    const __bf16* qrow = qbh + (size_t)(q0 + ln) * HDIM;
#pragma unroll
    for (int s = 0; s < 4; ++s)
      qf[s] = ld_frag(qrow + s * 32 + koff, qrow + s * 32 + koff + 8);
  }

  const v8f vzero = {0.f, 0.f, 0.f, 0.f, 0.f, 0.f, 0.f, 0.f};
  v8f oacc[8];  // O^T: 8 d-chunks of 16, N = query(lane)
#pragma unroll
  for (int dt = 0; dt < 8; ++dt) oacc[dt] = vzero;

  float mrun = -1e30f, lrun = 0.f;
  const float scale = 0.088388347648318447f;  // 1/sqrt(128)
  const int qglob = q0 + ln;
  const int kvend = q0 + 16;  // exclusive causal bound for this wave

  for (int kv = 0; kv < kvend; kv += 32) {
    // ---- scores S^T tiles (M=key, N=query), 16 keys per subtile ----
    v8f st[2];
#pragma unroll
    for (int c2 = 0; c2 < 2; ++c2) {
      v8f sacc = vzero;
      const __bf16* krow = kbh + (size_t)(kv + c2 * 16 + ln) * HDIM;
#pragma unroll
      for (int s = 0; s < 4; ++s) {
        v16bf af = ld_frag(krow + s * 32 + kb_off, krow + s * 32 + kb_off + 8);
        sacc = wmma_bf16(af, qf[s], sacc);
      }
      st[c2] = sacc;
    }

    // ---- online softmax (keys are the in-lane elements + lane^16 partner) --
    float sv[16];
    float smax = -1e30f;
#pragma unroll
    for (int c2 = 0; c2 < 2; ++c2) {
#pragma unroll
      for (int i = 0; i < 8; ++i) {
        int key = kv + c2 * 16 + i + 8 * hi;
        float s = st[c2][i] * scale;
        if (key > qglob) s = -1e30f;
        sv[c2 * 8 + i] = s;
        smax = fmaxf(smax, s);
      }
    }
    smax = fmaxf(smax, __shfl_xor(smax, 16, 32));
    float mnew  = fmaxf(mrun, smax);
    float alpha = __expf(mrun - mnew);

    float lsum = 0.f;
    unsigned* prow = (unsigned*)(pl + ln * 40);
#pragma unroll
    for (int c2 = 0; c2 < 2; ++c2) {
#pragma unroll
      for (int i = 0; i < 8; i += 2) {
        float p0 = __expf(sv[c2 * 8 + i] - mnew);
        float p1 = __expf(sv[c2 * 8 + i + 1] - mnew);
        lsum += p0 + p1;
        int key = c2 * 16 + i + 8 * hi;  // 0..31 within chunk
        prow[key >> 1] = pack2_bf16(p0, p1);
      }
    }
    lsum += __shfl_xor(lsum, 16, 32);
    lrun = lrun * alpha + lsum;
    mrun = mnew;

#pragma unroll
    for (int dt = 0; dt < 8; ++dt)
#pragma unroll
      for (int i = 0; i < 8; ++i) oacc[dt][i] *= alpha;

    __builtin_amdgcn_wave_barrier();  // keep LDS P stores before the reload

    // P^T B-fragment: lane -> query ln, element j -> key koff+j
    v16bf pf = ld_frag(pl + ln * 40 + koff, pl + ln * 40 + koff + 8);

    // ---- O^T += V^T * P^T over 8 d-chunks ----
#pragma unroll
    for (int dt = 0; dt < 8; ++dt) {
      const __bf16* vrow = vbh + (size_t)(dt * 16 + ln) * TSEQ + kv + kb_off;
      v16bf vf = ld_frag(vrow, vrow + 8);
      oacc[dt] = wmma_bf16(vf, pf, oacc[dt]);
    }
    __builtin_amdgcn_wave_barrier();
  }

  // ---- epilogue: normalize and store y[b][t][h*128+d] as bf16 ----
  float rl = 1.f / lrun;
  int b = bh >> 4, h = bh & (NHEAD - 1);
  size_t ybase = ((size_t)(b * TSEQ + q0 + ln)) * CDIM + h * HDIM;
#pragma unroll
  for (int dt = 0; dt < 8; ++dt) {
#pragma unroll
    for (int i = 0; i < 8; ++i) {
      int d = dt * 16 + i + 8 * hi;
      yb[ybase + d] = (__bf16)(oacc[dt][i] * rl);
    }
  }
}

// ---------------------------------------------------------------------------
// Launcher
// ---------------------------------------------------------------------------
extern "C" void kernel_launch(void* const* d_in, const int* in_sizes, int n_in,
                              void* d_out, int out_size, void* d_ws, size_t ws_size,
                              hipStream_t stream) {
  (void)in_sizes; (void)n_in; (void)out_size; (void)ws_size;
  const float* x     = (const float*)d_in[0];
  // d_in[1] = mask (causal handled analytically)
  const float* Wqkv  = (const float*)d_in[2];
  const float* Wproj = (const float*)d_in[3];
  const float* bproj = (const float*)d_in[4];
  float* out = (float*)d_out;

  char* ws = (char*)d_ws;
  size_t off = 0;
  auto alloc = [&](size_t bytes) -> void* {
    void* p = ws + off;
    off += (bytes + 255) & ~(size_t)255;
    return p;
  };
  __bf16* xbf    = (__bf16*)alloc((size_t)MROWS * CDIM * 2);        // 16 MB
  __bf16* WqkvT  = (__bf16*)alloc((size_t)3 * CDIM * CDIM * 2);     // 24 MB
  __bf16* WprojT = (__bf16*)alloc((size_t)CDIM * CDIM * 2);         //  8 MB
  __bf16* qb     = (__bf16*)alloc((size_t)32 * TSEQ * HDIM * 2);    // 16 MB
  __bf16* kbuf   = (__bf16*)alloc((size_t)32 * TSEQ * HDIM * 2);    // 16 MB
  __bf16* vt     = (__bf16*)alloc((size_t)32 * HDIM * TSEQ * 2);    // 16 MB
  __bf16* yb     = (__bf16*)alloc((size_t)MROWS * CDIM * 2);        // 16 MB

  // 1) convert inputs
  cvt_bf16_kernel<<<(MROWS * CDIM) / 256, 256, 0, stream>>>(x, xbf, MROWS * CDIM);
  transpose_cvt_kernel<<<dim3(3 * CDIM / 32, CDIM / 32), dim3(32, 8), 0, stream>>>(
      Wqkv, WqkvT, CDIM, 3 * CDIM);
  transpose_cvt_kernel<<<dim3(CDIM / 32, CDIM / 32), dim3(32, 8), 0, stream>>>(
      Wproj, WprojT, CDIM, CDIM);

  // 2) QKV projection, scatter into q/k/vT head layouts
  gemm_bf16_kernel<<<dim3(3 * CDIM / 256, MROWS / 128), 256, 0, stream>>>(
      xbf, WqkvT, MROWS, 3 * CDIM, CDIM, 0, qb, kbuf, vt, nullptr, nullptr);

  // 3) causal flash attention
  attn_kernel<<<dim3(TSEQ / 128, 32), 256, 0, stream>>>(qb, kbuf, vt, yb);

  // 4) output projection + bias -> f32
  gemm_bf16_kernel<<<dim3(CDIM / 256, MROWS / 128), 256, 0, stream>>>(
      yb, WprojT, MROWS, CDIM, CDIM, 1, nullptr, nullptr, nullptr, bproj, out);
}
